// NeuronCausalAttention_18038862643390
// MI455X (gfx1250) — matrix operand order repaired and checked by reading, not compiled, not run on hardware
//
#include <hip/hip_runtime.h>
#include <hip/hip_bf16.h>

// ---------------------------------------------------------------------------
// Types for CDNA5 WMMA (wave32): bf16 A/B fragments (16 bf16 = 8 VGPRs/lane),
// f32 C/D accumulator (8 VGPRs/lane).
// ---------------------------------------------------------------------------
typedef __attribute__((ext_vector_type(16))) __bf16 v16bf;
typedef __attribute__((ext_vector_type(8)))  __bf16 v8bf;
typedef __attribute__((ext_vector_type(8)))  float  v8f;

#define T_SEQ   384
#define N_NEUR  128
#define D_MODEL 512
#define N_HEADS 8
#define HEAD_D  64
#define M_TOT   (N_NEUR * T_SEQ)   // 49152 rows for the big GEMMs

__device__ __forceinline__ v8f wmma_bf16(v16bf a, v16bf b, v8f c) {
  // D = A(16x32) * B(32x16) + C, fp32 accumulate
  return __builtin_amdgcn_wmma_f32_16x16x32_bf16(false, a, false, b,
                                                 (short)0, c, false, false);
}

// Build an A-fragment (16x32 bf16, contiguous K per lane half) from two 16B chunks:
// per-lane K chunks [kb..kb+7] and [kb+8..kb+15], kb = (lane<16?0:16).
__device__ __forceinline__ v16bf make_afrag(const __bf16* p0, const __bf16* p1) {
  v8bf a = *(const v8bf*)p0;
  v8bf b = *(const v8bf*)p1;
  v16bf r;
#pragma unroll
  for (int i = 0; i < 8; ++i) { r[i] = a[i]; r[8 + i] = b[i]; }
  return r;
}

// ---------------------------------------------------------------------------
// 1) fp32 -> bf16 convert (weights)
// ---------------------------------------------------------------------------
__global__ void nca_cvt_bf16(const float* __restrict__ src,
                             __bf16* __restrict__ dst, int n) {
  int i = blockIdx.x * blockDim.x + threadIdx.x;
  if (i < n) dst[i] = (__bf16)src[i];
}

// ---------------------------------------------------------------------------
// 2) RMSNorm over D + transpose (B,T,N,D) -> xt[n][t][d] (bf16)
//    one 256-thread block per (t,n) row
// ---------------------------------------------------------------------------
__global__ void nca_rmsnorm(const float* __restrict__ x,
                            const float* __restrict__ w,
                            __bf16* __restrict__ xt) {
  int row = blockIdx.x;                 // row = t*128 + n
  int t = row >> 7, n = row & 127;
  const float* xr = x + (size_t)row * D_MODEL;
  float v0 = xr[threadIdx.x];
  float v1 = xr[threadIdx.x + 256];
  float s = v0 * v0 + v1 * v1;
#pragma unroll
  for (int off = 16; off; off >>= 1) s += __shfl_xor(s, off, 32);
  __shared__ float red[8];
  if ((threadIdx.x & 31) == 0) red[threadIdx.x >> 5] = s;
  __syncthreads();
  float tot = red[0] + red[1] + red[2] + red[3] + red[4] + red[5] + red[6] + red[7];
  float rs = rsqrtf(tot * (1.0f / D_MODEL) + 1e-6f);
  size_t ob = ((size_t)n * T_SEQ + t) * D_MODEL;
  xt[ob + threadIdx.x]       = (__bf16)(v0 * rs * w[threadIdx.x]);
  xt[ob + threadIdx.x + 256] = (__bf16)(v1 * rs * w[threadIdx.x + 256]);
}

// ---------------------------------------------------------------------------
// 3/6) GEMM  C[M x 512] = A[M x 512] @ W^T, W stored [n][k] (k contiguous).
//     Wave tile: 16(M) x 64(N), 4 f32 accumulators; K-loop step 32 (bf16 WMMA).
//     Block = 8 waves -> 128(M) x 64(N). Grid (M/128, 512/64).
//     mode 0: store bf16 to out_bf[m][n]
//     mode 1: store f32 to d_out at (t, r, n) with m = r*T + t
// ---------------------------------------------------------------------------
__global__ void nca_gemm(const __bf16* __restrict__ A,
                         const __bf16* __restrict__ W,
                         __bf16* __restrict__ out_bf,
                         float* __restrict__ out_f32,
                         int mode) {
  int lane = threadIdx.x & 31;
  int wave = threadIdx.x >> 5;
  int half = lane >> 4;        // which 16-lane half
  int l15  = lane & 15;
  int m0 = blockIdx.x * 128 + wave * 16;
  int n0 = blockIdx.y * 64;

  v8f acc[4] = {};
  int kbA = half ? 16 : 0;     // A-frag K sub-base per half
  int kbB = half ? 16 : 0;     // B-frag K sub-base per half

  for (int k0 = 0; k0 < D_MODEL; k0 += 32) {
    const __bf16* ap = A + (size_t)(m0 + l15) * D_MODEL + k0 + kbA;
    v16bf af = make_afrag(ap, ap + 8);
#pragma unroll
    for (int nt = 0; nt < 4; ++nt) {
      const __bf16* bp = W + (size_t)(n0 + nt * 16 + l15) * D_MODEL + k0 + kbB;
      v16bf bf_ = *(const v16bf*)bp;   // 16 consecutive K for this N column
      acc[nt] = wmma_bf16(af, bf_, acc[nt]);
    }
  }

#pragma unroll
  for (int nt = 0; nt < 4; ++nt) {
#pragma unroll
    for (int v = 0; v < 8; ++v) {
      int m = m0 + v + (half ? 8 : 0);     // C layout: row = v / v+8
      int n = n0 + nt * 16 + l15;
      float val = acc[nt][v];
      if (mode == 0) {
        out_bf[(size_t)m * D_MODEL + n] = (__bf16)val;
      } else {
        int r = m / T_SEQ, t = m % T_SEQ;  // scatter back to (B,T,N,D)
        out_f32[((size_t)t * N_NEUR + r) * D_MODEL + n] = val;
      }
    }
  }
}

// ---------------------------------------------------------------------------
// 4) RoPE in place on q and k (bf16). rotary_dim = 32 -> 16 pairs per head.
// ---------------------------------------------------------------------------
__global__ void nca_rope(__bf16* __restrict__ q, __bf16* __restrict__ k) {
  const size_t per = (size_t)N_NEUR * N_HEADS * T_SEQ * 16;
  size_t gid = (size_t)blockIdx.x * blockDim.x + threadIdx.x;
  __bf16* base = (gid < per) ? q : k;
  size_t g = (gid < per) ? gid : gid - per;
  int j = (int)(g & 15);  g >>= 4;
  int t = (int)(g % T_SEQ); g /= T_SEQ;
  int h = (int)(g & 7);   g >>= 3;
  int r = (int)g;
  size_t idx = ((size_t)r * T_SEQ + t) * D_MODEL + h * HEAD_D + 2 * j;
  float a = (float)base[idx], b = (float)base[idx + 1];
  float ang = (float)t * __powf(10000.0f, -(float)j / 16.0f);
  float sn = __sinf(ang), cs = __cosf(ang);
  base[idx]     = (__bf16)(a * cs - b * sn);
  base[idx + 1] = (__bf16)(a * sn + b * cs);
}

// ---------------------------------------------------------------------------
// 5) Causal flash attention per (r, h). Block = 4 waves x 16 queries = 64 q.
//    Key loop in steps of 32: k/v tiles staged in LDS; scores = 2x2 WMMA,
//    online softmax (f32), P re-layout via per-wave LDS, out += P @ V (4 WMMA).
//    Epilogue: /l, neuron mask, + residual(xt), store bf16 into y.
// ---------------------------------------------------------------------------
__global__ void nca_attn(const __bf16* __restrict__ q,
                         const __bf16* __restrict__ k,
                         const __bf16* __restrict__ v,
                         const __bf16* __restrict__ xt,
                         const int* __restrict__ pad,
                         __bf16* __restrict__ y) {
  int tb = blockIdx.x, h = blockIdx.y, r = blockIdx.z;
  int lane = threadIdx.x & 31;
  int wave = threadIdx.x >> 5;   // 0..3
  int half = lane >> 4, l15 = lane & 15;
  int q0 = tb * 64 + wave * 16;

  const size_t headoff = (size_t)r * T_SEQ * D_MODEL + (size_t)h * HEAD_D;
  const __bf16* qb = q + headoff;
  const __bf16* kb = k + headoff;
  const __bf16* vb = v + headoff;

  __shared__ alignas(64) __bf16 ktile[32][HEAD_D];   // [s][dh]
  __shared__ alignas(64) __bf16 vtile[HEAD_D][32];   // [dh][s] (transposed)
  __shared__ alignas(64) __bf16 pbuf[4][16][32];     // per-wave P re-layout

  // q fragments (held in registers): A layout, K = dh (two 32-chunks)
  int kbA = half ? 16 : 0;
  const __bf16* qp = qb + (size_t)(q0 + l15) * D_MODEL + kbA;
  v16bf qf0 = make_afrag(qp,      qp + 8);
  v16bf qf1 = make_afrag(qp + 32, qp + 40);

  v8f oacc[4] = {};
  float mrow[8], lrow[8];
#pragma unroll
  for (int i = 0; i < 8; ++i) { mrow[i] = -3.0e38f; lrow[i] = 0.0f; }

  const float scale = 0.125f;    // 1/sqrt(64)
  int sEnd = tb * 64 + 63;

  for (int s0 = 0; s0 <= sEnd; s0 += 32) {
    __syncthreads();
    {   // cooperative tile loads: 128 threads, 16 elems each per tile
      int row = threadIdx.x >> 2;       // 0..31
      int seg = threadIdx.x & 3;        // 0..3 (16 dh each)
      const __bf16* src = kb + (size_t)(s0 + row) * D_MODEL + seg * 16;
      *(v8bf*)&ktile[row][seg * 16]     = *(const v8bf*)src;
      *(v8bf*)&ktile[row][seg * 16 + 8] = *(const v8bf*)(src + 8);
      const __bf16* vsrc = vb + (size_t)(s0 + row) * D_MODEL + seg * 16;
#pragma unroll
      for (int e = 0; e < 16; ++e) vtile[seg * 16 + e][row] = vsrc[e];
    }
    __syncthreads();

    if (s0 <= q0 + 15) {                 // wave has at least one unmasked key
      // ---- scores: 16 x 32, K = 64 via two chained WMMA per 16-col subtile
      v8f sc[2];
#pragma unroll
      for (int snt = 0; snt < 2; ++snt) {
        int srow = snt * 16 + l15;       // B-frag N -> key index
        int dhb  = half ? 16 : 0;        // B-frag K sub-base
        v16bf kf0 = *(const v16bf*)&ktile[srow][dhb];
        v16bf kf1 = *(const v16bf*)&ktile[srow][dhb + 32];
        v8f a = {};
        a = wmma_bf16(qf0, kf0, a);
        a = wmma_bf16(qf1, kf1, a);
        sc[snt] = a;
      }
      // ---- scale + causal mask
#pragma unroll
      for (int snt = 0; snt < 2; ++snt)
#pragma unroll
        for (int vv = 0; vv < 8; ++vv) {
          int qg = q0 + vv + (half ? 8 : 0);
          int sg = s0 + snt * 16 + l15;
          float xsc = sc[snt][vv] * scale;
          sc[snt][vv] = (sg <= qg) ? xsc : -3.0e38f;
        }
      // ---- online softmax: row stats over 32 cols (16-lane butterflies)
      float alpha[8];
#pragma unroll
      for (int vv = 0; vv < 8; ++vv) {
        float t = fmaxf(sc[0][vv], sc[1][vv]);
        t = fmaxf(t, __shfl_xor(t, 1, 32));
        t = fmaxf(t, __shfl_xor(t, 2, 32));
        t = fmaxf(t, __shfl_xor(t, 4, 32));
        t = fmaxf(t, __shfl_xor(t, 8, 32));
        float mn = fmaxf(mrow[vv], t);
        alpha[vv] = __expf(mrow[vv] - mn);
        mrow[vv] = mn;
      }
#pragma unroll
      for (int snt = 0; snt < 2; ++snt)
#pragma unroll
        for (int vv = 0; vv < 8; ++vv)
          sc[snt][vv] = __expf(sc[snt][vv] - mrow[vv]);
#pragma unroll
      for (int vv = 0; vv < 8; ++vv) {
        float t = sc[0][vv] + sc[1][vv];
        t += __shfl_xor(t, 1, 32);
        t += __shfl_xor(t, 2, 32);
        t += __shfl_xor(t, 4, 32);
        t += __shfl_xor(t, 8, 32);
        lrow[vv] = lrow[vv] * alpha[vv] + t;
      }
#pragma unroll
      for (int nt = 0; nt < 4; ++nt)
#pragma unroll
        for (int vv = 0; vv < 8; ++vv) oacc[nt][vv] *= alpha[vv];

      // ---- P: C-layout -> A-layout via per-wave LDS buffer
#pragma unroll
      for (int snt = 0; snt < 2; ++snt)
#pragma unroll
        for (int vv = 0; vv < 8; ++vv)
          pbuf[wave][vv + (half ? 8 : 0)][snt * 16 + l15] = (__bf16)sc[snt][vv];
      asm volatile("s_wait_dscnt 0" ::: "memory");
      const __bf16* pp = &pbuf[wave][l15][0] + kbA;
      v16bf pf = make_afrag(pp, pp + 8);

      // ---- out += P(16x32) @ V(32x64): 4 WMMA over dh subtiles
#pragma unroll
      for (int nt = 0; nt < 4; ++nt) {
        int dh  = nt * 16 + l15;         // B-frag N -> dh
        int skb = half ? 16 : 0;         // B-frag K -> key index sub-base
        v16bf vf = *(const v16bf*)&vtile[dh][skb];
        oacc[nt] = wmma_bf16(pf, vf, oacc[nt]);
      }
    }
  }

  // ---- epilogue: normalize, neuron mask, residual, store bf16
  float maskv = (pad[r] != 0) ? 1.0f : 0.0f;
#pragma unroll
  for (int nt = 0; nt < 4; ++nt)
#pragma unroll
    for (int vv = 0; vv < 8; ++vv) {
      int t  = q0 + vv + (half ? 8 : 0);
      int dh = nt * 16 + l15;
      size_t idx = ((size_t)r * T_SEQ + t) * D_MODEL + h * HEAD_D + dh;
      float o = oacc[nt][vv] / lrow[vv];
      y[idx] = (__bf16)(maskv * o + (float)xt[idx]);
    }
}

// ---------------------------------------------------------------------------
// Host-side launcher
// ---------------------------------------------------------------------------
extern "C" void kernel_launch(void* const* d_in, const int* in_sizes, int n_in,
                              void* d_out, int out_size, void* d_ws, size_t ws_size,
                              hipStream_t stream) {
  (void)in_sizes; (void)n_in; (void)out_size; (void)ws_size;
  const float* x      = (const float*)d_in[0];
  const int*   pad    = (const int*)d_in[1];
  const float* norm_w = (const float*)d_in[2];
  const float* wq     = (const float*)d_in[3];
  const float* wk     = (const float*)d_in[4];
  const float* wv     = (const float*)d_in[5];
  const float* wo     = (const float*)d_in[6];
  float* out = (float*)d_out;

  char* ws = (char*)d_ws;
  const size_t WSZ  = (size_t)D_MODEL * D_MODEL * sizeof(__bf16);   // 512 KB
  const size_t XSZ  = (size_t)M_TOT * D_MODEL * sizeof(__bf16);     // 48 MB
  __bf16* wq_b = (__bf16*)(ws);
  __bf16* wk_b = (__bf16*)(ws + WSZ);
  __bf16* wv_b = (__bf16*)(ws + 2 * WSZ);
  __bf16* wo_b = (__bf16*)(ws + 3 * WSZ);
  __bf16* xt_b = (__bf16*)(ws + 4 * WSZ);
  __bf16* q_b  = (__bf16*)(ws + 4 * WSZ + XSZ);
  __bf16* k_b  = (__bf16*)(ws + 4 * WSZ + 2 * XSZ);
  __bf16* v_b  = (__bf16*)(ws + 4 * WSZ + 3 * XSZ);
  __bf16* y_b  = (__bf16*)(ws + 4 * WSZ + 4 * XSZ);

  // 1) weights -> bf16 (layout [n][k] kept: it is the WMMA B-fragment order)
  int wn = D_MODEL * D_MODEL;
  nca_cvt_bf16<<<(wn + 255) / 256, 256, 0, stream>>>(wq, wq_b, wn);
  nca_cvt_bf16<<<(wn + 255) / 256, 256, 0, stream>>>(wk, wk_b, wn);
  nca_cvt_bf16<<<(wn + 255) / 256, 256, 0, stream>>>(wv, wv_b, wn);
  nca_cvt_bf16<<<(wn + 255) / 256, 256, 0, stream>>>(wo, wo_b, wn);

  // 2) RMSNorm + transpose
  nca_rmsnorm<<<T_SEQ * N_NEUR, 256, 0, stream>>>(x, norm_w, xt_b);

  // 3) q/k/v projections (bf16 WMMA, f32 accumulate)
  dim3 ggrid(M_TOT / 128, D_MODEL / 64);
  nca_gemm<<<ggrid, 256, 0, stream>>>(xt_b, wq_b, q_b, nullptr, 0);
  nca_gemm<<<ggrid, 256, 0, stream>>>(xt_b, wk_b, k_b, nullptr, 0);
  nca_gemm<<<ggrid, 256, 0, stream>>>(xt_b, wv_b, v_b, nullptr, 0);

  // 4) RoPE on q and k (in place)
  size_t rope_threads = 2ull * N_NEUR * N_HEADS * T_SEQ * 16;
  nca_rope<<<(unsigned)(rope_threads / 256), 256, 0, stream>>>(q_b, k_b);

  // 5) causal attention + mask + residual -> y (bf16)
  dim3 agrid(T_SEQ / 64, N_HEADS, N_NEUR);
  nca_attn<<<agrid, 128, 0, stream>>>(q_b, k_b, v_b, xt_b, pad, y_b);

  // 6) output projection, scatter fp32 to (B,T,N,D)
  nca_gemm<<<ggrid, 256, 0, stream>>>(y_b, wo_b, nullptr, out, 1);
}
